// GroupMixAttention_45938970198161
// MI455X (gfx1250) — matrix operand order repaired and checked by reading, not compiled, not run on hardware
//
#include <hip/hip_runtime.h>
#include <hip/hip_bf16.h>

typedef __attribute__((ext_vector_type(16))) _Float16 v16h;
typedef __attribute__((ext_vector_type(8)))  _Float16 v8h;
typedef __attribute__((ext_vector_type(8)))  float    v8f;

#define B_  16
#define C_  256
#define G_  4
#define GD  64
#define N_  1024
#define LDK 72   // padded LDS row stride (halves): conflict-free, 16B-aligned rows

__device__ __forceinline__ v8f wmma_f16(v16h a, v16h b, v8f c) {
    return __builtin_amdgcn_wmma_f32_16x16x32_f16(false, a, false, b, (short)0, c, false, false);
}

// Async global->LDS 16B copy (CDNA5 GLOBAL_LOAD_ASYNC_TO_LDS_B128, ASYNCcnt).
// Generic pointers in the LDS aperture carry the LDS byte offset in bits [31:0].
__device__ __forceinline__ void async_g2l_b128(void* lds, const void* gptr) {
    unsigned off = (unsigned)(unsigned long long)lds;
    asm volatile("global_load_async_to_lds_b128 %0, %1, off"
                 :: "v"(off), "v"(gptr) : "memory");
}
__device__ __forceinline__ void wait_async0() {
    asm volatile("s_wait_asynccnt 0x0" ::: "memory");
}
__device__ __forceinline__ void wait_async4() {
    asm volatile("s_wait_asynccnt 0x4" ::: "memory");
}

// A operand (16 x 32 k-chunk) from row-major [M][K] f16 source, stride ld (halves).
// element (m = lane%16, k = (i&7) + 8*(lane/16) + 16*(i>>3)) -> two contiguous 16B runs.
__device__ __forceinline__ v16h frag_a_rm(const _Float16* p, int ld) {
    int lane = threadIdx.x & 31;
    const _Float16* row = p + (size_t)(lane & 15) * ld + 8 * (lane >> 4);
    v8h lo = *(const v8h*)(row);
    v8h hi = *(const v8h*)(row + 16);
    return __builtin_shufflevector(lo, hi, 0, 1, 2, 3, 4, 5, 6, 7,
                                           8, 9, 10, 11, 12, 13, 14, 15);
}

// B operand (32 k-chunk x 16) from TRANSPOSED [N][K] f16 source, stride ld (halves).
// element (k = (i&7) + 8*(lane/16) + 16*(i>>3), n = lane%16) -> two contiguous 16B runs.
__device__ __forceinline__ v16h frag_b_t(const _Float16* p, int ld) {
    int lane = threadIdx.x & 31;
    const _Float16* q = p + (size_t)(lane & 15) * ld + 8 * (lane >> 4);
    v8h lo = *(const v8h*)(q);
    v8h hi = *(const v8h*)(q + 16);
    return __builtin_shufflevector(lo, hi, 0, 1, 2, 3, 4, 5, 6, 7,
                                           8, 9, 10, 11, 12, 13, 14, 15);
}

// B operand from TRANSPOSED [N][K] f32 source (converted to f16); contiguous floats.
__device__ __forceinline__ v16h frag_b_t_f32(const float* p, int ld) {
    int lane = threadIdx.x & 31;
    const float* q = p + (size_t)(lane & 15) * ld + 8 * (lane >> 4);
    v16h b;
#pragma unroll
    for (int i = 0; i < 16; ++i) b[i] = (_Float16)q[(i & 7) + 16 * (i >> 3)];
    return b;
}

// Repack one 16x16 f32 D-tile (WMMA C/D layout) into f16 rows via per-wave LDS,
// then store each row half as one 16B global store. Tb: [16][24] per-wave buffer.
__device__ __forceinline__ void store_tile_f16(_Float16* dstRow0, size_t rowStride,
                                               const v8f& cacc,
                                               _Float16 (*Tb)[24]) {
    int lane = threadIdx.x & 31;
    int col = lane & 15, hi = lane >> 4;
#pragma unroll
    for (int r = 0; r < 8; ++r) Tb[r + 8 * hi][col] = (_Float16)cacc[r];
    __builtin_amdgcn_wave_barrier();
    int rr = lane >> 1, half = (lane & 1) * 8;
    v8h v = *(const v8h*)&Tb[rr][half];
    *(v8h*)(dstRow0 + (size_t)rr * rowStride + half) = v;
    __builtin_amdgcn_wave_barrier();
}

// -------------------- Kernel 1: QKV projections --------------------
// grid (B*G, N/128), block 256 (8 waves). Outputs (f16 workspace):
//   qt[bg][n][d] = scale * Q^T     kt[bg][n][d] = K^T     vd[bg][d][n] = V
__global__ __launch_bounds__(256) void qkv_kernel(const float* __restrict__ x,
                                                  const float* __restrict__ wq,
                                                  const float* __restrict__ wk,
                                                  const float* __restrict__ wv,
                                                  _Float16* __restrict__ qt,
                                                  _Float16* __restrict__ kt,
                                                  _Float16* __restrict__ vd) {
    __shared__ __align__(16) _Float16 Ws[3][GD][GD];      // row-major [d][c], 24 KB
    __shared__ __align__(16) _Float16 Xt[128][LDK];       // X^T [n][c], 18 KB
    __shared__ __align__(16) _Float16 Tb[8][16][24];      // per-wave repack, 6 KB

    const int tid = threadIdx.x;
    const int bg = blockIdx.x;
    const int b = bg / G_, g = bg % G_;
    const int n0 = blockIdx.y * 128;

    for (int idx = tid; idx < 3 * GD * GD / 4; idx += 256) {
        int mat = idx >> 10, r4 = (idx & 1023) * 4;
        const float* wsrc = (mat == 0) ? wq : ((mat == 1) ? wk : wv);
        float4 v = *(const float4*)&wsrc[g * GD * GD + r4];
        float s = (mat == 0) ? 0.125f : 1.0f;     // 1/sqrt(gd) folded into Q
        int row = r4 >> 6, c0 = r4 & 63;
        Ws[mat][row][c0 + 0] = (_Float16)(v.x * s);
        Ws[mat][row][c0 + 1] = (_Float16)(v.y * s);
        Ws[mat][row][c0 + 2] = (_Float16)(v.z * s);
        Ws[mat][row][c0 + 3] = (_Float16)(v.w * s);
    }
    for (int idx = tid; idx < GD * 32; idx += 256) {      // 64 rows x 32 float4s
        int c = idx >> 5, j4 = (idx & 31) * 4;
        float4 v = *(const float4*)&x[(((size_t)b * C_ + g * GD + c) << 10) + n0 + j4];
        Xt[j4 + 0][c] = (_Float16)v.x;
        Xt[j4 + 1][c] = (_Float16)v.y;
        Xt[j4 + 2][c] = (_Float16)v.z;
        Xt[j4 + 3][c] = (_Float16)v.w;
    }
    __syncthreads();

    const int wave = tid >> 5;
    const int nt = wave * 16;

#pragma unroll
    for (int dt = 0; dt < 4; ++dt) {
        const int d0 = dt * 16;
        v8f cq = {}, ck = {}, cv = {};
#pragma unroll
        for (int kc = 0; kc < GD; kc += 32) {
            v16h aX = frag_a_rm(&Xt[nt][kc], LDK);         // A = X^T [n x c]
            v16h aW = frag_a_rm(&Ws[2][d0][kc], GD);       // A = Wv  [d x c]
            v16h bQ = frag_b_t(&Ws[0][d0][kc], GD);        // B = Wq^T (src [d][c])
            v16h bK = frag_b_t(&Ws[1][d0][kc], GD);        // B = Wk^T (src [d][c])
            v16h bX = frag_b_t(&Xt[nt][kc], LDK);          // B = X    (src [n][c])
            cq = wmma_f16(aX, bQ, cq);
            ck = wmma_f16(aX, bK, ck);
            cv = wmma_f16(aW, bX, cv);
        }
        store_tile_f16(qt + ((size_t)bg * N_ + n0 + nt) * GD + d0, GD, cq, Tb[wave]);
        store_tile_f16(kt + ((size_t)bg * N_ + n0 + nt) * GD + d0, GD, ck, Tb[wave]);
        store_tile_f16(vd + ((size_t)bg * GD + d0) * N_ + n0 + nt, N_, cv, Tb[wave]);
    }
}

// -------------------- Kernel 2: flash attention --------------------
// grid (B*G, N/128), block 256 (8 waves, one 16-query tile each).
// Double-buffered K/V chunks streamed with CDNA5 async global->LDS loads:
// per-wave issue order is Q(4), c0(4), c1(4), ...; async loads complete in
// order, so after issuing chunk i+1, s_wait_asynccnt 4 guarantees chunk i
// (and Q) are resident while chunk i+1 is still in flight.
__global__ __launch_bounds__(256) void attn_kernel(const _Float16* __restrict__ qt,
                                                   const _Float16* __restrict__ kt,
                                                   const _Float16* __restrict__ vd,
                                                   _Float16* __restrict__ ot) {
    __shared__ __align__(16) _Float16 Qs[128][LDK];       // Q^T [n][c], 18 KB
    __shared__ __align__(16) _Float16 Kb[2][64][LDK];     // K^T chunks [m][c], 18 KB
    __shared__ __align__(16) _Float16 Vb[2][GD][LDK];     // V chunks  [d][m], 18 KB
    __shared__ __align__(16) _Float16 Ps[8][16][LDK];     // per-wave P [n][m], 18 KB

    const int tid = threadIdx.x;
    const int bg = blockIdx.x;
    const int b = bg / G_, g = bg % G_;
    const int n0 = blockIdx.y * 128;
    const int wave = tid >> 5, lane = tid & 31;
    const int col = lane & 15, hi = lane >> 4;

    // Async-stage the whole Q tile (4 ops/thread).
    for (int idx = tid; idx < 128 * GD / 8; idx += 256) {
        int n = idx >> 3, c8 = (idx & 7) * 8;
        async_g2l_b128(&Qs[n][c8], &qt[((size_t)bg * N_ + n0 + n) * GD + c8]);
    }
    // Prologue: chunk 0 into buffer 0 (4 ops/thread: 2 K + 2 V).
    for (int idx = tid; idx < 64 * 64 / 8; idx += 256) {
        int r = idx >> 3, c8 = (idx & 7) * 8;
        async_g2l_b128(&Kb[0][r][c8], &kt[((size_t)bg * N_ + r) * GD + c8]);
        async_g2l_b128(&Vb[0][r][c8], &vd[((size_t)bg * GD + r) * N_ + c8]);
    }

    v8f acc[4] = {{}, {}, {}, {}};
    float rowm[8], rowl[8];
#pragma unroll
    for (int r = 0; r < 8; ++r) { rowm[r] = -1e30f; rowl[r] = 0.0f; }

    for (int mc = 0; mc < N_ / 64; ++mc) {
        const int cur = mc & 1;
        if (mc + 1 < N_ / 64) {
            const int m1 = (mc + 1) * 64;
            for (int idx = tid; idx < 64 * 64 / 8; idx += 256) {
                int r = idx >> 3, c8 = (idx & 7) * 8;
                async_g2l_b128(&Kb[cur ^ 1][r][c8],
                               &kt[((size_t)bg * N_ + m1 + r) * GD + c8]);
                async_g2l_b128(&Vb[cur ^ 1][r][c8],
                               &vd[((size_t)bg * GD + r) * N_ + m1 + c8]);
            }
            wait_async4();   // chunk mc resident; chunk mc+1 still streaming
        } else {
            wait_async0();
        }
        __syncthreads();

        // S = Q^T K  (16 queries x 64 keys), four 16-wide subtiles
        v8f sc[4];
#pragma unroll
        for (int mt = 0; mt < 4; ++mt) {
            v8f s = {};
#pragma unroll
            for (int kc = 0; kc < GD; kc += 32) {
                v16h a = frag_a_rm(&Qs[wave * 16][kc], LDK);
                v16h bb = frag_b_t(&Kb[cur][mt * 16][kc], LDK);
                s = wmma_f16(a, bb, s);
            }
            sc[mt] = s;
        }

        // Online softmax: rows n = r + 8*hi live across each 16-lane half.
        float scl[8];
#pragma unroll
        for (int r = 0; r < 8; ++r) {
            float v = sc[0][r];
#pragma unroll
            for (int mt = 1; mt < 4; ++mt) v = fmaxf(v, sc[mt][r]);
#pragma unroll
            for (int off = 1; off < 16; off <<= 1) v = fmaxf(v, __shfl_xor(v, off, 32));
            float nm = fmaxf(rowm[r], v);
            scl[r] = __expf(rowm[r] - nm);
            rowm[r] = nm;
            float rs = 0.0f;
#pragma unroll
            for (int mt = 0; mt < 4; ++mt) {
                float p = __expf(sc[mt][r] - nm);
                rs += p;
                Ps[wave][r + 8 * hi][mt * 16 + col] = (_Float16)p;
            }
#pragma unroll
            for (int off = 1; off < 16; off <<= 1) rs += __shfl_xor(rs, off, 32);
            rowl[r] = rowl[r] * scl[r] + rs;
        }
#pragma unroll
        for (int dt = 0; dt < 4; ++dt)
#pragma unroll
            for (int r = 0; r < 8; ++r) acc[dt][r] *= scl[r];

        __syncthreads();  // uniform: make per-wave P stores visible before reads

        // O^T += P @ V^T   (16 queries x 64 dims)
#pragma unroll
        for (int dt = 0; dt < 4; ++dt) {
#pragma unroll
            for (int kc = 0; kc < 64; kc += 32) {
                v16h a = frag_a_rm(&Ps[wave][0][kc], LDK);
                v16h bb = frag_b_t(&Vb[cur][dt * 16][kc], LDK);
                acc[dt] = wmma_f16(a, bb, acc[dt]);
            }
        }
        __syncthreads();  // all waves done with Kb/Vb[cur] before it is refilled
    }

    // Normalize, repack through per-wave LDS, store O^T rows vectorized.
    float inv[8];
#pragma unroll
    for (int r = 0; r < 8; ++r) inv[r] = 1.0f / rowl[r];
#pragma unroll
    for (int dt = 0; dt < 4; ++dt)
#pragma unroll
        for (int r = 0; r < 8; ++r)
            Ps[wave][r + 8 * hi][dt * 16 + col] = (_Float16)(acc[dt][r] * inv[r]);
    __builtin_amdgcn_wave_barrier();
    {
        int rr = lane >> 1, h32 = (lane & 1) * 32;
        _Float16* orow = ot + ((size_t)b * N_ + n0 + wave * 16 + rr) * C_ + g * GD + h32;
#pragma unroll
        for (int t = 0; t < 4; ++t)
            *(v8h*)(orow + t * 8) = *(const v8h*)&Ps[wave][rr][h32 + t * 8];
    }
}

// -------------------- Kernel 3: output projection --------------------
// Y[b][co][n] = sum_c Wo[co][c] * O[b][c][n], O given as ot[b][n][c].
// Y^T tiles = O^T * Wo^T; transpose through LDS for contiguous fp32 stores.
__global__ __launch_bounds__(256) void proj_kernel(const _Float16* __restrict__ ot,
                                                   const float* __restrict__ wo,
                                                   float* __restrict__ y) {
    __shared__ float T[8][16][17];

    const int tid = threadIdx.x;
    const int b = blockIdx.x;
    const int co0 = blockIdx.y * 16;
    const int wave = tid >> 5, lane = tid & 31;
    const int col = lane & 15, hi = lane >> 4;
    const int n0 = blockIdx.z * 128 + wave * 16;

    v8f acc = {};
#pragma unroll
    for (int kc = 0; kc < C_; kc += 32) {
        v16h a = frag_a_rm(ot + ((size_t)b * N_ + n0) * C_ + kc, C_); // A = O^T [n x c]
        v16h bb = frag_b_t_f32(wo + (size_t)co0 * C_ + kc, C_);       // B = Wo^T (src [co][c])
        acc = wmma_f16(a, bb, acc);
    }

#pragma unroll
    for (int r = 0; r < 8; ++r) T[wave][r + 8 * hi][col] = acc[r];
    __syncthreads();
#pragma unroll
    for (int j = 0; j < 8; ++j) {
        int flat = j * 32 + lane;
        int co = flat >> 4, n = flat & 15;
        y[((size_t)b * C_ + co0 + co) * N_ + n0 + n] = T[wave][n][co];
    }
}

extern "C" void kernel_launch(void* const* d_in, const int* in_sizes, int n_in,
                              void* d_out, int out_size, void* d_ws, size_t ws_size,
                              hipStream_t stream) {
    const float* x  = (const float*)d_in[0];
    const float* wq = (const float*)d_in[1];
    const float* wk = (const float*)d_in[2];
    const float* wv = (const float*)d_in[3];
    const float* wo = (const float*)d_in[4];
    float* y = (float*)d_out;

    _Float16* ws = (_Float16*)d_ws;
    const size_t SZ = (size_t)B_ * G_ * N_ * GD;  // 4 Mi halves per buffer
    _Float16* qt = ws;
    _Float16* kt = ws + SZ;
    _Float16* vd = ws + 2 * SZ;
    _Float16* ot = ws + 3 * SZ;

    qkv_kernel<<<dim3(B_ * G_, N_ / 128), 256, 0, stream>>>(x, wq, wk, wv, qt, kt, vd);
    attn_kernel<<<dim3(B_ * G_, N_ / 128), 256, 0, stream>>>(qt, kt, vd, ot);
    proj_kernel<<<dim3(B_, C_ / 16, N_ / 128), 256, 0, stream>>>(ot, wo, y);
}
